// SuperHybridAttentionModel_87754771792440
// MI455X (gfx1250) — matrix operand order repaired and checked by reading, not compiled, not run on hardware
//
#include <hip/hip_runtime.h>
#include <cstdint>

#define B_     4
#define S_     4096
#define H_     2048
#define HC_    1024
#define HEADS_ 16
#define DH_    64
#define T_     512
#define NW_    8
#define MAXREL_ 128

typedef __bf16 bf16_t;
typedef __bf16 bf16x16 __attribute__((ext_vector_type(16)));
typedef float  f32x8   __attribute__((ext_vector_type(8)));

__device__ __forceinline__ void store_out(float* p, float v)  { *p = v; }
__device__ __forceinline__ void store_out(bf16_t* p, float v) { *p = (bf16_t)v; }

__device__ __forceinline__ f32x8 f32x8_zero() {
  f32x8 z;
#pragma unroll
  for (int i = 0; i < 8; ++i) z[i] = 0.0f;
  return z;
}

union FragU { bf16x16 v; uint4 q[2]; };

// Build a 16-element bf16 fragment from two 16-byte LDS chunks.
__device__ __forceinline__ bf16x16 frag_2x16B(const bf16_t* p0, const bf16_t* p1) {
  FragU u;
  u.q[0] = *(const uint4*)p0;
  u.q[1] = *(const uint4*)p1;
  return u.v;
}

// One 16-byte async global->LDS copy (CDNA5, tracked by ASYNCcnt).
// LDS address: low 32 bits of the generic pointer.
__device__ __forceinline__ void async_copy_b128(bf16_t* lds_dst, const bf16_t* gsrc) {
  uint32_t lds = (uint32_t)(uintptr_t)lds_dst;
  asm volatile("global_load_async_to_lds_b128 %0, %1, off"
               :: "v"(lds), "v"(gsrc) : "memory");
}

// ---------------------------------------------------------------------------
// Prep kernels: fp32 -> bf16 convert, and transpose-convert for weights.
// ---------------------------------------------------------------------------
__global__ __launch_bounds__(256)
void cvt_f32_bf16(const float* __restrict__ src, bf16_t* __restrict__ dst) {
  size_t i = ((size_t)blockIdx.x * 256 + threadIdx.x) * 4;
  float4 f = *(const float4*)(src + i);
  dst[i + 0] = (bf16_t)f.x;
  dst[i + 1] = (bf16_t)f.y;
  dst[i + 2] = (bf16_t)f.z;
  dst[i + 3] = (bf16_t)f.w;
}

// Wt[n][k] = (bf16)W[k][n];  W is K x N row-major.  grid: (N/32, K/32)
__global__ __launch_bounds__(256)
void transpose_cvt(const float* __restrict__ W, bf16_t* __restrict__ Wt, int K, int N) {
  __shared__ float t[32][33];
  int kt = blockIdx.y * 32, nt = blockIdx.x * 32;
  int tx = threadIdx.x & 31, ty = threadIdx.x >> 5;   // 32 x 8
#pragma unroll
  for (int i = 0; i < 32; i += 8)
    t[ty + i][tx] = W[(size_t)(kt + ty + i) * N + nt + tx];
  __syncthreads();
#pragma unroll
  for (int i = 0; i < 32; i += 8)
    Wt[(size_t)(nt + ty + i) * K + kt + tx] = (bf16_t)t[tx][ty + i];
}

// ---------------------------------------------------------------------------
// GEMM with bias: C[M,N] = A[M,K] * Bt[N,K]^T + bias[N]
// A row-major bf16, Bt pre-transposed bf16 [n][k], C bf16 or fp32.
// 128x128x32 tiles, double-buffered LDS filled by async global->LDS copies.
// 8 waves; each wave computes 32(M) x 64(N) via 2x4 WMMA tiles.
// ---------------------------------------------------------------------------
template <typename TOut>
__global__ __launch_bounds__(256)
void gemm_bias_wmma(const bf16_t* __restrict__ A, const bf16_t* __restrict__ Bt,
                    const float* __restrict__ bias, TOut* __restrict__ C,
                    int M, int N, int K)
{
  constexpr int BM = 128, BN = 128, BK = 32, LD = 40;  // row = 80 B (16B aligned)
  __shared__ bf16_t As[2][BM][LD];
  __shared__ bf16_t Bs[2][BN][LD];

  const int tid  = threadIdx.x;
  const int lane = tid & 31;
  const int wave = tid >> 5;
  const int wm   = (wave >> 1) * 32;
  const int wn   = (wave & 1) * 64;
  const int half = lane >> 4;
  const int l16  = lane & 15;

  const int m0 = blockIdx.y * BM;
  const int n0 = blockIdx.x * BN;

  const bf16_t* Abase = A  + (size_t)m0 * K;
  const bf16_t* Bbase = Bt + (size_t)n0 * K;

  // tile fill: 128 rows x 64 B; 4 lanes/row x 16 B, two issues of 64 rows
  const int rr = tid >> 2;         // 0..63
  const int c8 = (tid & 3) * 8;    // element offset of this lane's 16B chunk

  auto issue_tile = [&](int p, int kt) {
#pragma unroll
    for (int j = 0; j < 2; ++j) {
      int r = j * 64 + rr;
      async_copy_b128(&As[p][r][c8], Abase + (size_t)r * K + kt + c8);
    }
#pragma unroll
    for (int j = 0; j < 2; ++j) {
      int r = j * 64 + rr;
      async_copy_b128(&Bs[p][r][c8], Bbase + (size_t)r * K + kt + c8);
    }
  };

  f32x8 acc[2][4];
#pragma unroll
  for (int i = 0; i < 2; ++i)
#pragma unroll
    for (int j = 0; j < 4; ++j) acc[i][j] = f32x8_zero();

  const int nk = K / BK;
  issue_tile(0, 0);                        // prologue: 4 async ops outstanding

  for (int it = 0; it < nk; ++it) {
    const int p = it & 1;
    if (it + 1 < nk) {
      issue_tile(p ^ 1, (it + 1) * BK);    // prefetch next tile (8 outstanding)
      asm volatile("s_wait_asynccnt 4" ::: "memory");   // oldest tile landed
    } else {
      asm volatile("s_wait_asynccnt 0" ::: "memory");
    }
    __syncthreads();                       // tile visible to all waves

    bf16x16 afrag[2];
#pragma unroll
    for (int mt = 0; mt < 2; ++mt) {
      const bf16_t* row = &As[p][wm + mt * 16 + l16][0];
      afrag[mt] = frag_2x16B(row + half * 8, row + 16 + half * 8);
    }
    bf16x16 bfrag[4];
#pragma unroll
    for (int nt = 0; nt < 4; ++nt) {
      const bf16_t* q = &Bs[p][wn + nt * 16 + l16][0];
      bfrag[nt] = frag_2x16B(q + half * 8, q + 16 + half * 8);
    }
#pragma unroll
    for (int mt = 0; mt < 2; ++mt)
#pragma unroll
      for (int nt = 0; nt < 4; ++nt)
        acc[mt][nt] = __builtin_amdgcn_wmma_f32_16x16x32_bf16(
            false, afrag[mt], false, bfrag[nt], (short)0, acc[mt][nt], false, false);

    __syncthreads();                       // done reading before buffer reuse
  }

  // epilogue: C layout (lane<16: N=l16, rows half*8+i), + bias
#pragma unroll
  for (int mt = 0; mt < 2; ++mt)
#pragma unroll
    for (int nt = 0; nt < 4; ++nt) {
      int   gn = n0 + wn + nt * 16 + l16;
      float bv = bias[gn];
#pragma unroll
      for (int i = 0; i < 8; ++i) {
        int gm = m0 + wm + mt * 16 + half * 8 + i;
        store_out(&C[(size_t)gm * N + gn], acc[mt][nt][i] + bv);
      }
    }
}

// ---------------------------------------------------------------------------
// Windowed attention with relative bias (flash-style online softmax).
// grid.x = B*NW*HEADS, grid.y = T/128. Block: 256 threads (8 waves).
// ---------------------------------------------------------------------------
__global__ __launch_bounds__(256)
void attn_window(const bf16_t* __restrict__ q, const bf16_t* __restrict__ k,
                 const bf16_t* __restrict__ v, const float* __restrict__ rel_bias,
                 bf16_t* __restrict__ ctx)
{
  constexpr int BQ = 128, BKV = 128, LDQ = 72, LDK = 72, LDV = 136, LDP = 136;
  __shared__ bf16_t Qs[BQ][LDQ];
  __shared__ bf16_t Ks[BKV][LDK];
  __shared__ bf16_t Vt[DH_][LDV];
  __shared__ bf16_t Ps[BQ][LDP];
  __shared__ float  rbs[2 * MAXREL_ + 1];

  const int tid  = threadIdx.x;
  const int lane = tid & 31;
  const int wave = tid >> 5;
  const int half = lane >> 4;
  const int l16  = lane & 15;

  const int bh   = blockIdx.x;
  const int head = bh % HEADS_;
  const int w    = (bh / HEADS_) % NW_;
  const int b    = bh / (HEADS_ * NW_);
  const int q0   = blockIdx.y * BQ;

  const bf16_t* qg = q + ((size_t)b * S_ + (size_t)w * T_ + q0) * HC_ + head * DH_;
  const bf16_t* kg = k + ((size_t)b * S_ + (size_t)w * T_) * HC_ + head * DH_;
  const bf16_t* vg = v + ((size_t)b * S_ + (size_t)w * T_) * HC_ + head * DH_;

  for (int i = tid; i < 2 * MAXREL_ + 1; i += 256) rbs[i] = rel_bias[i];

  {
    int r0 = tid >> 4;
    int c0 = (tid & 15) * 4;
#pragma unroll
    for (int rp = 0; rp < 8; ++rp) {
      int r = r0 + rp * 16;
      const bf16_t* src = qg + (size_t)r * HC_ + c0;
#pragma unroll
      for (int j = 0; j < 4; ++j) Qs[r][c0 + j] = src[j];
    }
  }

  f32x8 o[4];
#pragma unroll
  for (int dt = 0; dt < 4; ++dt) o[dt] = f32x8_zero();
  float m_run[8], l_run[8];
#pragma unroll
  for (int i = 0; i < 8; ++i) { m_run[i] = -1e30f; l_run[i] = 0.0f; }

  const int qrow_base = q0 + wave * 16 + half * 8;

  for (int kt = 0; kt < T_; kt += BKV) {
    __syncthreads();
    {
      int r0 = tid >> 4;
      int c0 = (tid & 15) * 4;
#pragma unroll
      for (int rp = 0; rp < 8; ++rp) {
        int r = r0 + rp * 16;
        const bf16_t* ks = kg + (size_t)(kt + r) * HC_ + c0;
#pragma unroll
        for (int j = 0; j < 4; ++j) Ks[r][c0 + j] = ks[j];
        const bf16_t* vs = vg + (size_t)(kt + r) * HC_ + c0;
#pragma unroll
        for (int j = 0; j < 4; ++j) Vt[c0 + j][r] = vs[j];
      }
    }
    __syncthreads();

    f32x8 sc[8];
#pragma unroll
    for (int nt = 0; nt < 8; ++nt) sc[nt] = f32x8_zero();
#pragma unroll
    for (int ks = 0; ks < 2; ++ks) {
      const bf16_t* ar = &Qs[wave * 16 + l16][ks * 32];
      bf16x16 afrag = frag_2x16B(ar + half * 8, ar + 16 + half * 8);
#pragma unroll
      for (int nt = 0; nt < 8; ++nt) {
        const bf16_t* bp = &Ks[nt * 16 + l16][ks * 32];
        bf16x16 bfrag = frag_2x16B(bp + half * 8, bp + 16 + half * 8);
        sc[nt] = __builtin_amdgcn_wmma_f32_16x16x32_bf16(
            false, afrag, false, bfrag, (short)0, sc[nt], false, false);
      }
    }

    float mnew[8];
#pragma unroll
    for (int i = 0; i < 8; ++i) mnew[i] = m_run[i];
#pragma unroll
    for (int nt = 0; nt < 8; ++nt) {
      int col = kt + nt * 16 + l16;
#pragma unroll
      for (int i = 0; i < 8; ++i) {
        int rel = col - (qrow_base + i);
        rel = rel < -MAXREL_ ? -MAXREL_ : (rel > MAXREL_ ? MAXREL_ : rel);
        float s = sc[nt][i] * 0.125f + rbs[rel + MAXREL_];
        sc[nt][i] = s;
        mnew[i] = fmaxf(mnew[i], s);
      }
    }
#pragma unroll
    for (int i = 0; i < 8; ++i) {
      float mv = mnew[i];
      mv = fmaxf(mv, __shfl_xor(mv, 1, 32));
      mv = fmaxf(mv, __shfl_xor(mv, 2, 32));
      mv = fmaxf(mv, __shfl_xor(mv, 4, 32));
      mv = fmaxf(mv, __shfl_xor(mv, 8, 32));
      mnew[i] = mv;
    }
    float escale[8], ladd[8];
#pragma unroll
    for (int i = 0; i < 8; ++i) { escale[i] = __expf(m_run[i] - mnew[i]); ladd[i] = 0.0f; }

#pragma unroll
    for (int nt = 0; nt < 8; ++nt) {
#pragma unroll
      for (int i = 0; i < 8; ++i) {
        float p = __expf(sc[nt][i] - mnew[i]);
        ladd[i] += p;
        Ps[wave * 16 + half * 8 + i][nt * 16 + l16] = (bf16_t)p;
      }
    }
#pragma unroll
    for (int i = 0; i < 8; ++i) {
      float lv = ladd[i];
      lv += __shfl_xor(lv, 1, 32);
      lv += __shfl_xor(lv, 2, 32);
      lv += __shfl_xor(lv, 4, 32);
      lv += __shfl_xor(lv, 8, 32);
      l_run[i] = l_run[i] * escale[i] + lv;
      m_run[i] = mnew[i];
    }
#pragma unroll
    for (int dt = 0; dt < 4; ++dt)
#pragma unroll
      for (int i = 0; i < 8; ++i) o[dt][i] *= escale[i];

    __syncthreads();

#pragma unroll
    for (int ks = 0; ks < 4; ++ks) {
      const bf16_t* ar = &Ps[wave * 16 + l16][ks * 32];
      bf16x16 afrag = frag_2x16B(ar + half * 8, ar + 16 + half * 8);
#pragma unroll
      for (int dt = 0; dt < 4; ++dt) {
        const bf16_t* bp = &Vt[dt * 16 + l16][ks * 32];
        bf16x16 bfrag = frag_2x16B(bp + half * 8, bp + 16 + half * 8);
        o[dt] = __builtin_amdgcn_wmma_f32_16x16x32_bf16(
            false, afrag, false, bfrag, (short)0, o[dt], false, false);
      }
    }
  }

  bf16_t* og = ctx + ((size_t)b * S_ + (size_t)w * T_ + q0 + wave * 16 + half * 8) * HC_
                   + head * DH_;
#pragma unroll
  for (int i = 0; i < 8; ++i) {
    float inv = 1.0f / l_run[i];
#pragma unroll
    for (int dt = 0; dt < 4; ++dt)
      og[(size_t)i * HC_ + dt * 16 + l16] = (bf16_t)(o[dt][i] * inv);
  }
}

// ---------------------------------------------------------------------------
extern "C" void kernel_launch(void* const* d_in, const int* in_sizes, int n_in,
                              void* d_out, int out_size, void* d_ws, size_t ws_size,
                              hipStream_t stream)
{
  (void)in_sizes; (void)n_in; (void)out_size; (void)ws_size;

  const float* hs = (const float*)d_in[0];
  const float* Wc = (const float*)d_in[1];
  const float* bc = (const float*)d_in[2];
  const float* Wq = (const float*)d_in[3];
  const float* bq = (const float*)d_in[4];
  const float* Wk = (const float*)d_in[5];
  const float* bk = (const float*)d_in[6];
  const float* Wv = (const float*)d_in[7];
  const float* bv = (const float*)d_in[8];
  const float* rb = (const float*)d_in[9];
  const float* Wd = (const float*)d_in[10];
  const float* bd = (const float*)d_in[11];
  float* out = (float*)d_out;

  const size_t nh = (size_t)B_ * S_ * H_;    // 33.5M
  const size_t nx = (size_t)B_ * S_ * HC_;   // 16.8M

  bf16_t* p   = (bf16_t*)d_ws;
  bf16_t* hsb = p; p += nh;
  bf16_t* xw  = p; p += nx;
  bf16_t* qw  = p; p += nx;
  bf16_t* kw  = p; p += nx;
  bf16_t* vw  = p; p += nx;
  bf16_t* cw  = p; p += nx;
  bf16_t* Wct = p; p += (size_t)H_ * HC_;
  bf16_t* Wqt = p; p += (size_t)HC_ * HC_;
  bf16_t* Wkt = p; p += (size_t)HC_ * HC_;
  bf16_t* Wvt = p; p += (size_t)HC_ * HC_;
  bf16_t* Wdt = p; p += (size_t)HC_ * H_;

  const int M = B_ * S_;
  dim3 blk(256);

  // one-shot precision/layout prep
  cvt_f32_bf16<<<dim3((unsigned)(nh / (256 * 4))), blk, 0, stream>>>(hs, hsb);
  transpose_cvt<<<dim3(HC_ / 32, H_ / 32), blk, 0, stream>>>(Wc, Wct, H_, HC_);
  transpose_cvt<<<dim3(HC_ / 32, HC_ / 32), blk, 0, stream>>>(Wq, Wqt, HC_, HC_);
  transpose_cvt<<<dim3(HC_ / 32, HC_ / 32), blk, 0, stream>>>(Wk, Wkt, HC_, HC_);
  transpose_cvt<<<dim3(HC_ / 32, HC_ / 32), blk, 0, stream>>>(Wv, Wvt, HC_, HC_);
  transpose_cvt<<<dim3(H_ / 32, HC_ / 32), blk, 0, stream>>>(Wd, Wdt, HC_, H_);

  // compress: x = hs @ Wc + bc
  gemm_bias_wmma<bf16_t><<<dim3(HC_ / 128, M / 128), blk, 0, stream>>>(
      hsb, Wct, bc, xw, M, HC_, H_);
  // q/k/v projections
  gemm_bias_wmma<bf16_t><<<dim3(HC_ / 128, M / 128), blk, 0, stream>>>(
      xw, Wqt, bq, qw, M, HC_, HC_);
  gemm_bias_wmma<bf16_t><<<dim3(HC_ / 128, M / 128), blk, 0, stream>>>(
      xw, Wkt, bk, kw, M, HC_, HC_);
  gemm_bias_wmma<bf16_t><<<dim3(HC_ / 128, M / 128), blk, 0, stream>>>(
      xw, Wvt, bv, vw, M, HC_, HC_);
  // windowed attention with relative bias
  attn_window<<<dim3(B_ * NW_ * HEADS_, T_ / 128), blk, 0, stream>>>(qw, kw, vw, rb, cw);
  // decompress: out = ctx @ Wd + bd (fp32 out)
  gemm_bias_wmma<float><<<dim3(H_ / 128, M / 128), blk, 0, stream>>>(
      cw, Wdt, bd, out, M, H_, HC_);
}
